// MultiHeadAttention_23175643529832
// MI455X (gfx1250) — compile-verified, matrix operand order checked by reading, not run on hardware
//
#include <hip/hip_runtime.h>

typedef __attribute__((ext_vector_type(16))) __bf16 v16bf;
typedef __attribute__((ext_vector_type(8)))  __bf16 v8bf;
typedef __attribute__((ext_vector_type(8)))  float  v8f;
typedef __attribute__((ext_vector_type(4)))  unsigned int v4u;
typedef __attribute__((ext_vector_type(8)))  int v8i;
typedef __attribute__((ext_vector_type(4)))  int v4i;

#define B_  2
#define S_  2048
#define D_  1024
#define H_  16
#define DH_ 64

// LDS row stride for TDM-staged tiles: 32 bf16 (64B) + 16B TDM pad = 80B = 40 elems
#define LDSTRIDE 40

static __device__ __forceinline__ unsigned short f2bf(float f) {
    union { float f; unsigned u; } c; c.f = f;
    unsigned u = c.u;
    u += 0x7FFFu + ((u >> 16) & 1u);   // round-to-nearest-even
    return (unsigned short)(u >> 16);
}

static __device__ __forceinline__ v8f wmma_bf16(v16bf a, v16bf b, v8f c) {
    return __builtin_amdgcn_wmma_f32_16x16x32_bf16(
        /*neg_a=*/false, a, /*neg_b=*/false, b,
        /*c_mod=*/(short)0, c, /*reuse_a=*/false, /*reuse_b=*/false);
}

union Frag { v16bf v; v8bf h[2]; };

// ---- Tensor Data Mover: load a 128-row x 32-elem bf16 tile (row stride =
// strideElems in memory) into LDS, packed 64B/row + 16B pad -> 80B LDS rows.
// D# layout per CDNA5 ISA ch.8 (group0: count/lds/global/type, group1: dims).
static __device__ __forceinline__ void tdm_load_tile_128x32(
    unsigned ldsByteAddr, const unsigned short* gsrc, int strideElems)
{
    unsigned long long ga = (unsigned long long)(uintptr_t)gsrc;
    v4u g0 = { 1u,                                   // count=1, is_restore=0
               ldsByteAddr,                          // lds_addr [63:32]
               (unsigned)ga,                         // global_addr low
               (unsigned)(ga >> 32) | (2u << 30) };  // global_addr hi | type=2
    v8i g1 = { (1 << 16)                             // data_size = 2 bytes
             | (1 << 20)                             // pad_enable
             | (3 << 22)                             // pad_interval: 16 DWORDs (64B)
             | (3 << 25),                            // pad_amount:   4 DWORDs (16B)
               0,                                    // barrier_addr=0, tdim0 lo16=0
               0x10,                                 // tensor_dim0 = 1<<20 (no OOB)
               0x10 | (32 << 16),                    // tensor_dim1 = 1<<20, tile_dim0=32
               128,                                  // tile_dim1 = 128, tile_dim2 = 0
               strideElems,                          // tensor_dim0_stride (low 32)
               0, 0 };
    v4i z4 = { 0, 0, 0, 0 };                         // groups 2/3 unused (2D tensor)
    v8i z8 = { 0, 0, 0, 0, 0, 0, 0, 0 };
    __builtin_amdgcn_tensor_load_to_lds(g0, g1, z4, z4, z8, 0);
}

// ---------------- fp32 -> bf16 conversion (+ weight transpose) ----------------
// Weights are transposed once so GEMM B-tiles are row-major [n][k] like A-tiles.
__global__ void cvt_kernel(const float* __restrict__ x, const float* __restrict__ wq,
                           const float* __restrict__ wo,
                           unsigned short* __restrict__ xb, unsigned short* __restrict__ wqt,
                           unsigned short* __restrict__ wot,
                           int nx, int nq, int no)
{
    int i  = blockIdx.x * blockDim.x + threadIdx.x;
    int st = gridDim.x * blockDim.x;
    for (int t = i; t < nx; t += st) xb[t] = f2bf(x[t]);
    for (int t = i; t < nq; t += st) {               // [D, 3D] -> [3D, D]
        int k = t / (3 * D_); int n = t - k * (3 * D_);
        wqt[(size_t)n * D_ + k] = f2bf(wq[t]);
    }
    for (int t = i; t < no; t += st) {               // [D, D] -> [D, D]^T
        int k = t >> 10; int n = t & (D_ - 1);
        wot[(size_t)n * D_ + k] = f2bf(wo[t]);
    }
}

// ---------------- tiled bf16 WMMA GEMM, TDM-staged tiles ----------------
// C[M,N] = A[M,K] * Bt[N,K]^T.  Block tile 128x128x32, 8 waves (2x4), wave 64x32.
// EPI==0: QKV epilogue (RoPE on Q/K, Q scaled 1/sqrt(dh), scatter Qh/Kh [bh,S,dh],
//         V transposed Vt [bh,dh,S], bf16).   EPI==1: acc + bias[n], fp32 store.
template<int EPI>
__global__ void __launch_bounds__(256) gemm_bf16_kernel(
    const unsigned short* __restrict__ A, const unsigned short* __restrict__ Bt,
    int M, int N, int K,
    unsigned short* __restrict__ Qh, unsigned short* __restrict__ Kh,
    unsigned short* __restrict__ Vt,
    const float* __restrict__ bias, float* __restrict__ Cout)
{
    __shared__ __align__(16) unsigned short As[128 * LDSTRIDE];
    __shared__ __align__(16) unsigned short Bs[128 * LDSTRIDE];

    const int tid   = threadIdx.x;
    const int wave  = tid >> 5, lane = tid & 31;
    const int lo    = lane & 15, hi = lane >> 4;
    const int waveM = wave >> 2, waveN = wave & 3;
    const int bm0   = blockIdx.y * 128;
    const int bn0   = blockIdx.x * 128;

    const unsigned asAddr = (unsigned)(unsigned long long)(uintptr_t)(void*)As;
    const unsigned bsAddr = (unsigned)(unsigned long long)(uintptr_t)(void*)Bs;

    v8f acc[4][2];
    #pragma unroll
    for (int i = 0; i < 4; i++)
        #pragma unroll
        for (int j = 0; j < 2; j++)
            #pragma unroll
            for (int e = 0; e < 8; e++) acc[i][j][e] = 0.0f;

    for (int k0 = 0; k0 < K; k0 += 32) {
        if (wave == 0) {   // one wave drives the TDM; descriptors are uniform
            tdm_load_tile_128x32(asAddr, A  + (size_t)bm0 * K + k0, K);
            tdm_load_tile_128x32(bsAddr, Bt + (size_t)bn0 * K + k0, K);
            __builtin_amdgcn_s_wait_tensorcnt(0);
        }
        __syncthreads();

        Frag af[4], bfr[2];
        #pragma unroll
        for (int mf = 0; mf < 4; mf++) {
            const unsigned short* p = As + (waveM * 64 + mf * 16 + lo) * LDSTRIDE;
            af[mf].h[0] = *(const v8bf*)(p + hi * 8);
            af[mf].h[1] = *(const v8bf*)(p + 16 + hi * 8);
        }
        #pragma unroll
        for (int nf = 0; nf < 2; nf++) {
            const unsigned short* p = Bs + (waveN * 32 + nf * 16 + lo) * LDSTRIDE;
            bfr[nf].h[0] = *(const v8bf*)(p + hi * 8);
            bfr[nf].h[1] = *(const v8bf*)(p + 16 + hi * 8);
        }
        #pragma unroll
        for (int mf = 0; mf < 4; mf++)
            #pragma unroll
            for (int nf = 0; nf < 2; nf++)
                acc[mf][nf] = wmma_bf16(af[mf].v, bfr[nf].v, acc[mf][nf]);
        __syncthreads();
    }

    // -------- epilogue --------
    #pragma unroll
    for (int mf = 0; mf < 4; mf++) {
        #pragma unroll
        for (int nf = 0; nf < 2; nf++) {
            v8f a = acc[mf][nf];
            const int gcol = bn0 + waveN * 32 + nf * 16 + lo;
            if (EPI == 0) {
                const int sec  = gcol >> 10;       // 0=Q 1=K 2=V
                const int dcol = gcol & 1023;
                const int hh   = dcol >> 6;
                const int d    = dcol & 63;
                const float fr  = __powf(10000.0f, -(float)(d & 31) * (1.0f / 32.0f));
                const float sgn = (d & 1) ? 1.0f : -1.0f;
                #pragma unroll
                for (int r = 0; r < 8; r++) {
                    const int grow = bm0 + waveM * 64 + mf * 16 + r + 8 * hi;
                    const int bb   = grow >> 11;
                    const int sp   = grow & (S_ - 1);
                    float xv = a[r];
                    float xp = __shfl_xor(xv, 1);  // interleaved RoPE pair = lane^1
                    float val;
                    if (sec < 2) {
                        float ang = (float)sp * fr;
                        float sn, cs;
                        __sincosf(ang, &sn, &cs);
                        val = xv * cs + sgn * xp * sn;
                        if (sec == 0) val *= 0.125f;   // 1/sqrt(64) folded into Q
                    } else {
                        val = xv;
                    }
                    const size_t bh = (size_t)bb * H_ + hh;
                    if (sec == 0)      Qh[(bh * S_ + sp) * DH_ + d] = f2bf(val);
                    else if (sec == 1) Kh[(bh * S_ + sp) * DH_ + d] = f2bf(val);
                    else               Vt[(bh * DH_ + d) * S_ + sp] = f2bf(val);
                }
            } else {
                const float bv = bias[gcol];
                #pragma unroll
                for (int r = 0; r < 8; r++) {
                    const int grow = bm0 + waveM * 64 + mf * 16 + r + 8 * hi;
                    Cout[(size_t)grow * N + gcol] = a[r] + bv;
                }
            }
        }
    }
}

// ---------------- flash attention (per-wave 16 query rows, K/V tile = 64) ----------------
__global__ void __launch_bounds__(256) attn_kernel(
    const unsigned short* __restrict__ Qh, const unsigned short* __restrict__ Kh,
    const unsigned short* __restrict__ Vt, unsigned short* __restrict__ Ah)
{
    __shared__ __align__(16) unsigned short Pl[8][16 * 64];  // per-wave P staging
    const int wave = threadIdx.x >> 5, lane = threadIdx.x & 31;
    const int lo = lane & 15, hi = lane >> 4;
    const int bh = blockIdx.y;
    const int b  = bh >> 4, h = bh & 15;
    const int q0 = blockIdx.x * 128 + wave * 16;

    const unsigned short* Qb = Qh + (size_t)bh * S_ * DH_;
    const unsigned short* Kb = Kh + (size_t)bh * S_ * DH_;
    const unsigned short* Vb = Vt + (size_t)bh * DH_ * S_;
    unsigned short* Pw = &Pl[wave][0];

    Frag qf[2];
    {
        const unsigned short* p = Qb + (size_t)(q0 + lo) * DH_;
        #pragma unroll
        for (int c = 0; c < 2; c++) {
            qf[c].h[0] = *(const v8bf*)(p + c * 32 + hi * 8);
            qf[c].h[1] = *(const v8bf*)(p + c * 32 + 16 + hi * 8);
        }
    }

    v8f o[4];
    float m_run[8], l_run[8];
    #pragma unroll
    for (int dt = 0; dt < 4; dt++)
        #pragma unroll
        for (int e = 0; e < 8; e++) o[dt][e] = 0.0f;
    #pragma unroll
    for (int r = 0; r < 8; r++) { m_run[r] = -3.402823466e38f; l_run[r] = 0.0f; }

    for (int kk0 = 0; kk0 < S_; kk0 += 64) {
        // ---- S = Q K^T (16 x 64) ----
        v8f sc[4];
        #pragma unroll
        for (int nt = 0; nt < 4; nt++) {
            Frag kf0, kf1;
            const unsigned short* p = Kb + (size_t)(kk0 + nt * 16 + lo) * DH_;
            kf0.h[0] = *(const v8bf*)(p + hi * 8);
            kf0.h[1] = *(const v8bf*)(p + 16 + hi * 8);
            kf1.h[0] = *(const v8bf*)(p + 32 + hi * 8);
            kf1.h[1] = *(const v8bf*)(p + 48 + hi * 8);
            v8f z;
            #pragma unroll
            for (int e = 0; e < 8; e++) z[e] = 0.0f;
            z      = wmma_bf16(qf[0].v, kf0.v, z);
            sc[nt] = wmma_bf16(qf[1].v, kf1.v, z);
        }
        // ---- online softmax over 64 keys ----
        float scl[8];
        #pragma unroll
        for (int r = 0; r < 8; r++) {
            float mx = fmaxf(fmaxf(sc[0][r], sc[1][r]), fmaxf(sc[2][r], sc[3][r]));
            #pragma unroll
            for (int m = 1; m < 16; m <<= 1) mx = fmaxf(mx, __shfl_xor(mx, m));
            float nm = fmaxf(m_run[r], mx);
            scl[r]   = __expf(m_run[r] - nm);
            m_run[r] = nm;
        }
        #pragma unroll
        for (int nt = 0; nt < 4; nt++)
            #pragma unroll
            for (int r = 0; r < 8; r++)
                sc[nt][r] = __expf(sc[nt][r] - m_run[r]);
        #pragma unroll
        for (int r = 0; r < 8; r++) {
            float sm = sc[0][r] + sc[1][r] + sc[2][r] + sc[3][r];
            #pragma unroll
            for (int m = 1; m < 16; m <<= 1) sm += __shfl_xor(sm, m);
            l_run[r] = l_run[r] * scl[r] + sm;
        }
        #pragma unroll
        for (int dt = 0; dt < 4; dt++)
            #pragma unroll
            for (int r = 0; r < 8; r++)
                o[dt][r] *= scl[r];
        // ---- C-layout -> A-layout transpose of P via per-wave LDS ----
        #pragma unroll
        for (int nt = 0; nt < 4; nt++)
            #pragma unroll
            for (int r = 0; r < 8; r++)
                Pw[(r + 8 * hi) * 64 + nt * 16 + lo] = f2bf(sc[nt][r]);
        Frag pf[2];
        #pragma unroll
        for (int c = 0; c < 2; c++) {
            const unsigned short* p = Pw + lo * 64;
            pf[c].h[0] = *(const v8bf*)(p + c * 32 + hi * 8);
            pf[c].h[1] = *(const v8bf*)(p + c * 32 + 16 + hi * 8);
        }
        // ---- O += P V ----
        #pragma unroll
        for (int dt = 0; dt < 4; dt++) {
            Frag vf0, vf1;
            const unsigned short* vp = Vb + (size_t)(dt * 16 + lo) * S_ + kk0;
            vf0.h[0] = *(const v8bf*)(vp + hi * 8);
            vf0.h[1] = *(const v8bf*)(vp + 16 + hi * 8);
            vf1.h[0] = *(const v8bf*)(vp + 32 + hi * 8);
            vf1.h[1] = *(const v8bf*)(vp + 48 + hi * 8);
            o[dt] = wmma_bf16(pf[0].v, vf0.v, o[dt]);
            o[dt] = wmma_bf16(pf[1].v, vf1.v, o[dt]);
        }
    }

    #pragma unroll
    for (int r = 0; r < 8; r++) l_run[r] = 1.0f / l_run[r];
    #pragma unroll
    for (int dt = 0; dt < 4; dt++)
        #pragma unroll
        for (int r = 0; r < 8; r++) {
            int row = q0 + r + 8 * hi;
            Ah[((size_t)b * S_ + row) * D_ + h * DH_ + dt * 16 + lo] = f2bf(o[dt][r] * l_run[r]);
        }
}

// ---------------- launch ----------------
extern "C" void kernel_launch(void* const* d_in, const int* in_sizes, int n_in,
                              void* d_out, int out_size, void* d_ws, size_t ws_size,
                              hipStream_t stream)
{
    const float* x    = (const float*)d_in[0];
    const float* Wqkv = (const float*)d_in[1];
    const float* Wout = (const float*)d_in[2];
    const float* bout = (const float*)d_in[3];
    float* out = (float*)d_out;

    char* ws = (char*)d_ws;
    unsigned short* xb  = (unsigned short*)(ws);                          // 8 MiB
    unsigned short* wqt = (unsigned short*)(ws + (size_t)8  * (1 << 20)); // 6 MiB [3D,D]
    unsigned short* wot = (unsigned short*)(ws + (size_t)14 * (1 << 20)); // 2 MiB [D,D]
    unsigned short* Qh  = (unsigned short*)(ws + (size_t)16 * (1 << 20)); // 8 MiB
    unsigned short* Kh  = (unsigned short*)(ws + (size_t)24 * (1 << 20)); // 8 MiB
    unsigned short* Vt  = (unsigned short*)(ws + (size_t)32 * (1 << 20)); // 8 MiB
    unsigned short* Ah  = (unsigned short*)(ws + (size_t)40 * (1 << 20)); // 8 MiB

    (void)in_sizes; (void)n_in; (void)out_size; (void)ws_size;

    cvt_kernel<<<2048, 256, 0, stream>>>(x, Wqkv, Wout, xb, wqt, wot,
                                         B_ * S_ * D_, D_ * 3 * D_, D_ * D_);

    gemm_bf16_kernel<0><<<dim3(3 * D_ / 128, B_ * S_ / 128), 256, 0, stream>>>(
        xb, wqt, B_ * S_, 3 * D_, D_, Qh, Kh, Vt, nullptr, nullptr);

    attn_kernel<<<dim3(S_ / 128, B_ * H_), 256, 0, stream>>>(Qh, Kh, Vt, Ah);

    gemm_bf16_kernel<1><<<dim3(D_ / 128, B_ * S_ / 128), 256, 0, stream>>>(
        Ah, wot, B_ * S_, D_, D_, nullptr, nullptr, nullptr, bout, out);
}
